// YiJingTransformerLayer_72705206387133
// MI455X (gfx1250) — matrix operand order repaired and checked by reading, not compiled, not run on hardware
//
#include <hip/hip_runtime.h>
#include <hip/hip_bf16.h>

// ---------------------------------------------------------------------------
// YiJing transformer layer for MI455X (gfx1250, wave32, WMMA bf16 path)
// B=4, T=2048, C=1024, H=8, D=128
// ---------------------------------------------------------------------------

#define Bn 4
#define Tn 2048
#define Cn 1024
#define Hn 8
#define Dh 128
#define MT (Bn * Tn) // 8192 rows

typedef __attribute__((ext_vector_type(16))) __bf16 v16bf;
typedef __attribute__((ext_vector_type(8)))  float  v8f;

union ABu { v16bf v; uint4 u[2]; };
union CFu { v8f   v; float f[8]; };

static __device__ __forceinline__ unsigned short f2bf(float x) {
    unsigned int u = __float_as_uint(x);
    u += 0x7FFFu + ((u >> 16) & 1u); // round-to-nearest-even
    return (unsigned short)(u >> 16);
}
static __device__ __forceinline__ float bf2f(unsigned short h) {
    return __uint_as_float(((unsigned int)h) << 16);
}
// reductions across the 16-lane half-wave (xor bits 0..3 stay in half)
static __device__ __forceinline__ float rmax16(float v) {
    v = fmaxf(v, __shfl_xor(v, 1, 32));
    v = fmaxf(v, __shfl_xor(v, 2, 32));
    v = fmaxf(v, __shfl_xor(v, 4, 32));
    v = fmaxf(v, __shfl_xor(v, 8, 32));
    return v;
}
static __device__ __forceinline__ float rsum16(float v) {
    v += __shfl_xor(v, 1, 32);
    v += __shfl_xor(v, 2, 32);
    v += __shfl_xor(v, 4, 32);
    v += __shfl_xor(v, 8, 32);
    return v;
}

// ---------------------------------------------------------------------------
// Kernel 1: fp32 -> bf16 weight conversion
// ---------------------------------------------------------------------------
__global__ __launch_bounds__(256) void cvt_kernel(const float* __restrict__ src,
                                                  unsigned short* __restrict__ dst,
                                                  int n) {
    for (int i = blockIdx.x * 256 + threadIdx.x; i < n; i += gridDim.x * 256)
        dst[i] = f2bf(src[i]);
}

// ---------------------------------------------------------------------------
// Kernel 2: LayerNorm -> bf16 xn  (one block per row of C=1024)
// ---------------------------------------------------------------------------
__global__ __launch_bounds__(256) void ln_kernel(const float* __restrict__ x,
                                                 const float* __restrict__ g,
                                                 const float* __restrict__ bta,
                                                 unsigned short* __restrict__ xn) {
    __shared__ float rs[256], rq[256];
    const int row = blockIdx.x;
    const int tid = threadIdx.x;
    float v[4];
    float s = 0.f, q = 0.f;
#pragma unroll
    for (int i = 0; i < 4; ++i) {
        v[i] = x[row * Cn + tid + i * 256];
        s += v[i];
        q += v[i] * v[i];
    }
    rs[tid] = s; rq[tid] = q;
    __syncthreads();
    for (int st = 128; st > 0; st >>= 1) {
        if (tid < st) { rs[tid] += rs[tid + st]; rq[tid] += rq[tid + st]; }
        __syncthreads();
    }
    const float mean = rs[0] * (1.0f / Cn);
    const float var  = rq[0] * (1.0f / Cn) - mean * mean;
    const float rstd = rsqrtf(var + 1e-5f);
#pragma unroll
    for (int i = 0; i < 4; ++i) {
        const int c = tid + i * 256;
        xn[row * Cn + c] = f2bf((v[i] - mean) * rstd * g[c] + bta[c]);
    }
}

// ---------------------------------------------------------------------------
// Kernel 3: QKV projection (bf16 WMMA) + RoPE epilogue via LDS strip.
// grid = (H, MT/16, 3) ; block = 256 (8 waves). Each wave: one 16x16 tile,
// the workgroup covers one 16-row x 128-col (one head) strip.
// mode 0 -> q (RoPE), 1 -> k (RoPE), 2 -> v (stored transposed [B,H,D,T])
// ---------------------------------------------------------------------------
__global__ __launch_bounds__(256) void qkv_kernel(const unsigned short* __restrict__ xn,
                                                  const unsigned short* __restrict__ wq,
                                                  const unsigned short* __restrict__ wk,
                                                  const unsigned short* __restrict__ wv,
                                                  unsigned short* __restrict__ qb,
                                                  unsigned short* __restrict__ kb,
                                                  unsigned short* __restrict__ vtb) {
    const int mode = blockIdx.z;
    const unsigned short* W = (mode == 0) ? wq : (mode == 1) ? wk : wv;
    const int h   = blockIdx.x;
    const int n0  = h * Dh;
    const int m0  = blockIdx.y * 16;
    const int wid = threadIdx.x >> 5;
    const int lane = threadIdx.x & 31;
    const int lo = lane & 15, hi = lane >> 4;

    const int ncol = n0 + wid * 16 + lo;   // B-fragment: this lane's column == W row
    const int arow = m0 + lo;              // A-fragment: this lane's xn row

    CFu acc;
#pragma unroll
    for (int j = 0; j < 8; ++j) acc.f[j] = 0.f;

    for (int k0 = 0; k0 < Cn; k0 += 32) {
        ABu a, bm;
        const unsigned short* ap = xn + arow * Cn + k0 + hi * 8;
        a.u[0] = *(const uint4*)ap;
        a.u[1] = *(const uint4*)(ap + 16);
        const unsigned short* bp = W + ncol * Cn + k0 + hi * 8;
        bm.u[0] = *(const uint4*)bp;
        bm.u[1] = *(const uint4*)(bp + 16);
        __builtin_prefetch(bp + 32, 0, 1); // global_prefetch_b8: next K block
        acc.v = __builtin_amdgcn_wmma_f32_16x16x32_bf16(
            false, a.v, false, bm.v, (short)0, acc.v, false, false);
    }

    __shared__ __attribute__((aligned(16))) float strip[16][Dh];
#pragma unroll
    for (int j = 0; j < 8; ++j)
        strip[j + 8 * hi][wid * 16 + lo] = acc.f[j];
    __syncthreads();

    const int b  = m0 / Tn;
    const int t0 = m0 % Tn;
    for (int i = threadIdx.x; i < 16 * Dh; i += 256) {
        const int r = i >> 7;
        const int d = i & (Dh - 1);
        const int t = t0 + r;
        const float xv = strip[r][d];
        if (mode == 2) {
            vtb[((b * Hn + h) * Dh + d) * Tn + t] = f2bf(xv);
        } else {
            const float x2 = strip[r][d ^ 64];
            const int dd = d & 63;
            // inv freq = 10000^(-2*dd/128); ln(10000) = 9.2103403720
            const float ang = (float)t * __expf(-(float)(2 * dd) * (9.2103403720f / 128.0f));
            float sn, cs;
            __sincosf(ang, &sn, &cs);
            const float rv = xv * cs + ((d < 64) ? -x2 : x2) * sn;
            unsigned short* dst = (mode == 0) ? qb : kb;
            dst[((b * Hn + h) * Tn + t) * Dh + d] = f2bf(rv);
        }
    }
}

// ---------------------------------------------------------------------------
// Kernel 4: fused causal flash attention (std + geo) per (b,h,qtile).
// One wave handles 16 query rows; 32-key blocks; 24 WMMAs per block.
// grid = B*H*(T/16)/8 blocks of 256 threads (8 waves).
// ---------------------------------------------------------------------------
__global__ __launch_bounds__(256) void attn_kernel(const unsigned short* __restrict__ qbuf,
                                                   const unsigned short* __restrict__ kbuf,
                                                   const unsigned short* __restrict__ vtb,
                                                   const float* __restrict__ head_scales,
                                                   const float* __restrict__ head_dirs,
                                                   unsigned short* __restrict__ attn_out) {
    __shared__ __attribute__((aligned(16))) unsigned short Ps[8][16 * 32];
    __shared__ __attribute__((aligned(16))) unsigned short Pg[8][16 * 32];

    const int wid  = threadIdx.x >> 5;
    const int lane = threadIdx.x & 31;
    const int lo = lane & 15, hi = lane >> 4;

    const int gid = blockIdx.x * 8 + wid;
    const int qi  = gid & (Tn / 16 - 1);
    const int bh  = gid >> 7;              // T/16 == 128
    const int h   = bh & (Hn - 1);
    const int b   = bh >> 3;

    const unsigned short* Q  = qbuf + (size_t)(b * Hn + h) * Tn * Dh;
    const unsigned short* K  = kbuf + (size_t)(b * Hn + h) * Tn * Dh;
    const unsigned short* VT = vtb  + (size_t)(b * Hn + h) * Dh * Tn;
    const int q0 = qi * 16;

    // resident Q fragments (16x128 -> 4 A-fragments of 16x32)
    ABu qf[4];
    {
        const unsigned short* qr = Q + (q0 + lo) * Dh;
#pragma unroll
        for (int kk = 0; kk < 4; ++kk) {
            const unsigned short* ap = qr + kk * 32 + hi * 8;
            qf[kk].u[0] = *(const uint4*)ap;
            qf[kk].u[1] = *(const uint4*)(ap + 16);
        }
    }

    // geo projections for this tile's query rows
    const float d0 = head_dirs[h * 3 + 0];
    const float d1 = head_dirs[h * 3 + 1];
    const float d2 = head_dirs[h * 3 + 2];
    float qp[8];
#pragma unroll
    for (int j = 0; j < 8; ++j) {
        const unsigned short* qr = Q + (q0 + j + 8 * hi) * Dh;
        qp[j] = bf2f(qr[0]) * d0 + bf2f(qr[1]) * d1 + bf2f(qr[2]) * d2;
    }

    CFu os[8], og[8];
    float ms[8], ls[8], mg[8], lg[8];
#pragma unroll
    for (int j = 0; j < 8; ++j) {
#pragma unroll
        for (int n = 0; n < 8; ++n) { /* placeholder */ }
        ms[j] = -3.0e38f; ls[j] = 0.f;
        mg[j] = -3.0e38f; lg[j] = 0.f;
    }
#pragma unroll
    for (int n = 0; n < 8; ++n)
#pragma unroll
        for (int j = 0; j < 8; ++j) { os[n].f[j] = 0.f; og[n].f[j] = 0.f; }

    const float scl = 0.08838834764831845f; // 1/sqrt(128)
    const int nblk = (qi + 2) >> 1;         // ceil((qi+1)*16 / 32)

    for (int blk = 0; blk < nblk; ++blk) {
        const int kb0 = blk * 32;

        // ---- scores: two 16x16 tiles over this 32-key block -------------
        CFu sc[2];
#pragma unroll
        for (int kt = 0; kt < 2; ++kt) {
#pragma unroll
            for (int j = 0; j < 8; ++j) sc[kt].f[j] = 0.f;
            const unsigned short* kr = K + (kb0 + kt * 16 + lo) * Dh;
#pragma unroll
            for (int kk = 0; kk < 4; ++kk) {
                ABu bm;
                const unsigned short* bp = kr + kk * 32 + hi * 8;
                bm.u[0] = *(const uint4*)bp;
                bm.u[1] = *(const uint4*)(bp + 16);
                sc[kt].v = __builtin_amdgcn_wmma_f32_16x16x32_bf16(
                    false, qf[kk].v, false, bm.v, (short)0, sc[kt].v, false, false);
            }
        }

        // geo key projections for this block
        float kp0, kp1;
        {
            const unsigned short* k0r = K + (kb0 + lo) * Dh;
            const unsigned short* k1r = K + (kb0 + 16 + lo) * Dh;
            kp0 = bf2f(k0r[0]) * d0 + bf2f(k0r[1]) * d1 + bf2f(k0r[2]) * d2;
            kp1 = bf2f(k1r[0]) * d0 + bf2f(k1r[1]) * d1 + bf2f(k1r[2]) * d2;
        }

        // ---- online softmax (std + geo), stage probs to LDS -------------
#pragma unroll
        for (int j = 0; j < 8; ++j) {
            const int row = q0 + j + 8 * hi;
            float a0 = sc[0].f[j] * scl;
            float a1 = sc[1].f[j] * scl;
            float g0 = qp[j] * kp0;
            float g1 = qp[j] * kp1;
            if (kb0 + lo      > row) { a0 = -3.0e38f; g0 = -3.0e38f; }
            if (kb0 + 16 + lo > row) { a1 = -3.0e38f; g1 = -3.0e38f; }

            // std
            {
                const float mn = fmaxf(ms[j], rmax16(fmaxf(a0, a1)));
                const float p0 = __expf(a0 - mn);
                const float p1 = __expf(a1 - mn);
                const float rs = rsum16(p0 + p1);
                const float ef = __expf(ms[j] - mn);
                ls[j] = ls[j] * ef + rs;
                ms[j] = mn;
#pragma unroll
                for (int n = 0; n < 8; ++n) os[n].f[j] *= ef;
                Ps[wid][(j + 8 * hi) * 32 + lo]      = f2bf(p0);
                Ps[wid][(j + 8 * hi) * 32 + 16 + lo] = f2bf(p1);
            }
            // geo
            {
                const float mn = fmaxf(mg[j], rmax16(fmaxf(g0, g1)));
                const float p0 = __expf(g0 - mn);
                const float p1 = __expf(g1 - mn);
                const float rs = rsum16(p0 + p1);
                const float ef = __expf(mg[j] - mn);
                lg[j] = lg[j] * ef + rs;
                mg[j] = mn;
#pragma unroll
                for (int n = 0; n < 8; ++n) og[n].f[j] *= ef;
                Pg[wid][(j + 8 * hi) * 32 + lo]      = f2bf(p0);
                Pg[wid][(j + 8 * hi) * 32 + 16 + lo] = f2bf(p1);
            }
        }

        // per-wave LDS staging: no barrier needed, but wait for DS writes
        asm volatile("s_wait_dscnt 0x0" ::: "memory");

        // re-read probs in A-fragment layout
        ABu pa, pgf;
        {
            const unsigned short* pr = &Ps[wid][lo * 32 + hi * 8];
            pa.u[0] = *(const uint4*)pr;
            pa.u[1] = *(const uint4*)(pr + 16);
            const unsigned short* pg = &Pg[wid][lo * 32 + hi * 8];
            pgf.u[0] = *(const uint4*)pg;
            pgf.u[1] = *(const uint4*)(pg + 16);
        }

        // ---- P @ V over the 32-key block (V stored transposed) ----------
#pragma unroll
        for (int n = 0; n < 8; ++n) {
            ABu bm;
            const unsigned short* vp = VT + (n * 16 + lo) * Tn + kb0 + hi * 8;
            bm.u[0] = *(const uint4*)vp;
            bm.u[1] = *(const uint4*)(vp + 16);
            os[n].v = __builtin_amdgcn_wmma_f32_16x16x32_bf16(
                false, pa.v, false, bm.v, (short)0, os[n].v, false, false);
            og[n].v = __builtin_amdgcn_wmma_f32_16x16x32_bf16(
                false, pgf.v, false, bm.v, (short)0, og[n].v, false, false);
        }
    }

    // ---- epilogue: normalize, blend geo, store bf16 [B*T, C] ------------
    const float hs = head_scales[h];
    float invs[8], invg[8];
#pragma unroll
    for (int j = 0; j < 8; ++j) { invs[j] = 1.0f / ls[j]; invg[j] = 1.0f / lg[j]; }
#pragma unroll
    for (int n = 0; n < 8; ++n) {
#pragma unroll
        for (int j = 0; j < 8; ++j) {
            const float o = os[n].f[j] * invs[j];
            const float gv = og[n].f[j] * invg[j];
            const float res = o + hs * (gv - o);
            const int t = q0 + j + 8 * hi;
            const int col = h * Dh + n * 16 + lo;
            attn_out[(size_t)(b * Tn + t) * Cn + col] = f2bf(res);
        }
    }
}

// ---------------------------------------------------------------------------
// Kernel 5: output projection (bf16 WMMA) + residual, fp32 out.
// grid = (C/128, MT/16); block = 256 (8 waves, one 16x16 tile each).
// ---------------------------------------------------------------------------
__global__ __launch_bounds__(256) void oproj_kernel(const unsigned short* __restrict__ attn,
                                                    const unsigned short* __restrict__ wo,
                                                    const float* __restrict__ x,
                                                    float* __restrict__ out) {
    const int n0  = blockIdx.x * 128;
    const int m0  = blockIdx.y * 16;
    const int wid = threadIdx.x >> 5;
    const int lane = threadIdx.x & 31;
    const int lo = lane & 15, hi = lane >> 4;

    const int ncol = n0 + wid * 16 + lo;
    const int arow = m0 + lo;

    CFu acc;
#pragma unroll
    for (int j = 0; j < 8; ++j) acc.f[j] = 0.f;

    for (int k0 = 0; k0 < Cn; k0 += 32) {
        ABu a, bm;
        const unsigned short* ap = attn + (size_t)arow * Cn + k0 + hi * 8;
        a.u[0] = *(const uint4*)ap;
        a.u[1] = *(const uint4*)(ap + 16);
        const unsigned short* bp = wo + (size_t)ncol * Cn + k0 + hi * 8;
        bm.u[0] = *(const uint4*)bp;
        bm.u[1] = *(const uint4*)(bp + 16);
        __builtin_prefetch(bp + 32, 0, 1);
        acc.v = __builtin_amdgcn_wmma_f32_16x16x32_bf16(
            false, a.v, false, bm.v, (short)0, acc.v, false, false);
    }

#pragma unroll
    for (int j = 0; j < 8; ++j) {
        const int m = m0 + j + 8 * hi;
        out[(size_t)m * Cn + ncol] = x[(size_t)m * Cn + ncol] + acc.f[j];
    }
}

// ---------------------------------------------------------------------------
// Host launcher
// ---------------------------------------------------------------------------
extern "C" void kernel_launch(void* const* d_in, const int* in_sizes, int n_in,
                              void* d_out, int out_size, void* d_ws, size_t ws_size,
                              hipStream_t stream) {
    const float* x   = (const float*)d_in[0];
    const float* Wq  = (const float*)d_in[1];
    const float* Wk  = (const float*)d_in[2];
    const float* Wv  = (const float*)d_in[3];
    const float* Wo  = (const float*)d_in[4];
    const float* lng = (const float*)d_in[5];
    const float* lnb = (const float*)d_in[6];
    const float* hsc = (const float*)d_in[7];
    const float* hdr = (const float*)d_in[8];
    float* out = (float*)d_out;

    // workspace layout (bytes)
    char* ws = (char*)d_ws;
    const size_t SZ_MTC = (size_t)MT * Cn * sizeof(unsigned short); // 16 MB
    const size_t SZ_W   = (size_t)Cn * Cn * sizeof(unsigned short); //  2 MB
    unsigned short* xn   = (unsigned short*)(ws);
    unsigned short* wqb  = (unsigned short*)(ws + SZ_MTC);
    unsigned short* wkb  = (unsigned short*)(ws + SZ_MTC + 1 * SZ_W);
    unsigned short* wvb  = (unsigned short*)(ws + SZ_MTC + 2 * SZ_W);
    unsigned short* wob  = (unsigned short*)(ws + SZ_MTC + 3 * SZ_W);
    unsigned short* qb   = (unsigned short*)(ws + SZ_MTC + 4 * SZ_W);
    unsigned short* kb   = (unsigned short*)(ws + 2 * SZ_MTC + 4 * SZ_W);
    unsigned short* vtb  = (unsigned short*)(ws + 3 * SZ_MTC + 4 * SZ_W);
    unsigned short* attn = (unsigned short*)(ws + 4 * SZ_MTC + 4 * SZ_W);

    const int nW = Cn * Cn;
    cvt_kernel<<<1024, 256, 0, stream>>>(Wq, wqb, nW);
    cvt_kernel<<<1024, 256, 0, stream>>>(Wk, wkb, nW);
    cvt_kernel<<<1024, 256, 0, stream>>>(Wv, wvb, nW);
    cvt_kernel<<<1024, 256, 0, stream>>>(Wo, wob, nW);

    ln_kernel<<<MT, 256, 0, stream>>>(x, lng, lnb, xn);

    qkv_kernel<<<dim3(Hn, MT / 16, 3), 256, 0, stream>>>(xn, wqb, wkb, wvb, qb, kb, vtb);

    // B*H*(T/16) wave tasks, 8 waves per block
    attn_kernel<<<(Bn * Hn * (Tn / 16)) / 8, 256, 0, stream>>>(qb, kb, vtb, hsc, hdr, attn);

    oproj_kernel<<<dim3(Cn / 128, MT / 16), 256, 0, stream>>>(attn, wob, x, out);

    (void)in_sizes; (void)n_in; (void)out_size; (void)ws_size;
}
